// XLSTMPatternCell_31387620999752
// MI455X (gfx1250) — matrix operand order repaired and checked by reading, not compiled, not run on hardware
//
#include <hip/hip_runtime.h>
#include <hip/hip_bf16.h>

typedef _Float16 v16h __attribute__((ext_vector_type(16)));
typedef _Float16 v8h  __attribute__((ext_vector_type(8)));
typedef _Float16 v4h  __attribute__((ext_vector_type(4)));
typedef float    v8f  __attribute__((ext_vector_type(8)));

#define B_   8
#define T_   4096
#define D_   256
#define U_   256
#define P_   6
#define M_   (B_ * T_)          // 32768 rows of x
#define MU_  ((size_t)M_ * U_)  // elements per projection

// LDS geometry (f16 elements). Row pad of 8 f16 (16 B) keeps b128 fragment
// loads 16B-aligned and rotates banks by 4 dwords per row -> conflict-free.
#define LROW 264
#define LDSX_ELEMS (128 * LROW)           // x tile: 128 rows
#define LDSW_ELEMS (256 * LROW)           // W[p]^T: 256 n-rows
#define LDS_BYTES  ((LDSX_ELEMS + LDSW_ELEMS) * 2)   // 202752 B

#define CAT16(lo, hi) __builtin_shufflevector((lo), (hi), \
    0,1,2,3,4,5,6,7,8,9,10,11,12,13,14,15)

// ---------------------------------------------------------------------------
// Phase 0a: x fp32 -> f16 (4 elements / thread)
// ---------------------------------------------------------------------------
__global__ __launch_bounds__(256) void cvt_x_f16(
    const float* __restrict__ x, _Float16* __restrict__ xh)
{
  const int g = blockIdx.x * blockDim.x + threadIdx.x;   // 0 .. M*D/4-1
  const float4 v = ((const float4*)x)[g];
  v4h h = { (_Float16)v.x, (_Float16)v.y, (_Float16)v.z, (_Float16)v.w };
  ((v4h*)xh)[g] = h;
}

// ---------------------------------------------------------------------------
// Phase 0b: W[p][k][n] fp32 -> Wt[p][n][k] f16 (transpose once, W is tiny)
// ---------------------------------------------------------------------------
__global__ __launch_bounds__(256) void cvt_wT_f16(
    const float* __restrict__ W, _Float16* __restrict__ Wt)
{
  const int g = blockIdx.x * blockDim.x + threadIdx.x;   // 0 .. 6*256*256-1
  const int p = g >> 16;
  const int r = g & 65535;
  const int k = r >> 8;
  const int n = r & 255;
  Wt[p * 65536 + n * 256 + k] = (_Float16)W[g];
}

// ---------------------------------------------------------------------------
// Phase 1: proj[p, m, u] = act_p( x[m,:] @ W[p,:,u] + b[p,u] )
// grid = (M/128, 6).  Block: 256 threads (8 waves).  Dynamic LDS holds the
// 128-row f16 x tile and the full 256x256 f16 W[p]^T.  Wave w keeps its
// entire A slice (8 x v16h) in registers, and computes 16 n-tiles in strips
// of 4 with software-pipelined B-fragment loads (next k-step's B fragments
// are issued before the current k-step's WMMAs).
// ---------------------------------------------------------------------------
__global__ __launch_bounds__(256) void xlstm_proj_wmma(
    const _Float16* __restrict__ xh,    // [M, D] f16
    const _Float16* __restrict__ Wt,    // [6, U, D] f16 (transposed)
    const float*    __restrict__ bias,  // [6, U]
    _Float16*       __restrict__ proj)  // [6, M, U] f16
{
  extern __shared__ __align__(16) _Float16 lds[];
  _Float16* ldsX = lds;                 // [128][LROW]
  _Float16* ldsW = lds + LDSX_ELEMS;    // [256][LROW]

  const int p   = blockIdx.y;
  const int m0  = blockIdx.x * 128;
  const int tid = threadIdx.x;

  // ---- Stage x tile: 128 rows x 256 f16, contiguous 16B chunks ----
  {
    const v8h* src = (const v8h*)(xh + (size_t)m0 * D_);   // 4096 chunks
    #pragma unroll
    for (int it = 0; it < 16; ++it) {
      const int o   = it * 256 + tid;      // chunk id
      const int row = o >> 5;              // 32 chunks per row
      const int col = o & 31;
      *(v8h*)&ldsX[row * LROW + col * 8] = src[o];
    }
  }
  // ---- Stage W[p]^T: 256 n-rows x 256 f16 ----
  {
    const v8h* src = (const v8h*)(Wt + (size_t)p * 65536); // 8192 chunks
    #pragma unroll
    for (int it = 0; it < 32; ++it) {
      const int o   = it * 256 + tid;
      const int row = o >> 5;
      const int col = o & 31;
      *(v8h*)&ldsW[row * LROW + col * 8] = src[o];
    }
  }
  __syncthreads();

  const int wave   = tid >> 5;
  const int lane   = tid & 31;
  const int hi     = lane >> 4;        // lane half selects K sub-range
  const int lane16 = lane & 15;

  // ---- Preload the wave's entire A slice into registers (8 x v16h) ----
  const _Float16* arow = ldsX + (wave * 16 + lane16) * LROW;
  v16h afrag[8];
  #pragma unroll
  for (int k8 = 0; k8 < 8; ++k8) {
    v8h lo  = *(const v8h*)&arow[k8 * 32 + 8 * hi];
    v8h hi8 = *(const v8h*)&arow[k8 * 32 + 16 + 8 * hi];
    afrag[k8] = CAT16(lo, hi8);
  }

  const _Float16* wrow = ldsW + lane16 * LROW + 8 * hi;
  auto loadB = [&](int nt, int k8) -> v16h {
    const _Float16* base = wrow + (size_t)nt * (16 * LROW) + k8 * 32;
    return CAT16(*(const v8h*)base, *(const v8h*)(base + 16));
  };

  const float* bp = bias + p * U_;

  #pragma unroll
  for (int s = 0; s < 4; ++s) {        // 4 strips of 4 n-tiles
    const int nt0 = s * 4;
    v8f acc0 = {}, acc1 = {}, acc2 = {}, acc3 = {};

    // Pipeline prologue: B fragments for k-step 0.
    v16h b0 = loadB(nt0 + 0, 0);
    v16h b1 = loadB(nt0 + 1, 0);
    v16h b2 = loadB(nt0 + 2, 0);
    v16h b3 = loadB(nt0 + 3, 0);

    #pragma unroll
    for (int k8 = 0; k8 < 8; ++k8) {
      v16h n0, n1, n2, n3;
      if (k8 < 7) {                    // issue next step's loads first
        n0 = loadB(nt0 + 0, k8 + 1);
        n1 = loadB(nt0 + 1, k8 + 1);
        n2 = loadB(nt0 + 2, k8 + 1);
        n3 = loadB(nt0 + 3, k8 + 1);
      }
      const v16h a = afrag[k8];
      acc0 = __builtin_amdgcn_wmma_f32_16x16x32_f16(false, a, false, b0,
                                                    (short)0, acc0, false, false);
      acc1 = __builtin_amdgcn_wmma_f32_16x16x32_f16(false, a, false, b1,
                                                    (short)0, acc1, false, false);
      acc2 = __builtin_amdgcn_wmma_f32_16x16x32_f16(false, a, false, b2,
                                                    (short)0, acc2, false, false);
      acc3 = __builtin_amdgcn_wmma_f32_16x16x32_f16(false, a, false, b3,
                                                    (short)0, acc3, false, false);
      if (k8 < 7) { b0 = n0; b1 = n1; b2 = n2; b3 = n3; }
    }

    // Bias + activation + f16 store. C layout: lane = N, vgpr r -> M = r+8*hi.
    v8f accs[4] = { acc0, acc1, acc2, acc3 };
    #pragma unroll
    for (int j = 0; j < 4; ++j) {
      const int ncol = (nt0 + j) * 16 + lane16;
      const float bi = bp[ncol];
      _Float16* orow = proj + (size_t)p * MU_
                            + (size_t)(m0 + wave * 16) * U_ + ncol;
      #pragma unroll
      for (int r = 0; r < 8; ++r) {
        float v = accs[j][r] + bi;
        if (p == 3)      v = __expf(v);                    // i = exp(zi)
        else if (p >= 4) v = 1.0f / (1.0f + __expf(-v));   // f, o = sigmoid
        orow[(size_t)(r + 8 * hi) * U_] = (_Float16)v;
      }
    }
  }
}

// ---------------------------------------------------------------------------
// Phase 2: sequential scan over T, one thread per (b,u) channel (2048 total,
// 64 single-wave workgroups). Loads are independent of the recurrence state,
// so unrolling exposes memory-level parallelism.
// ---------------------------------------------------------------------------
__global__ __launch_bounds__(32) void xlstm_scan(
    const _Float16* __restrict__ proj,  // [6, M, U] f16
    const float*    __restrict__ c0,
    const float*    __restrict__ n0v,
    float*          __restrict__ out)   // [B, T, U]
{
  const int idx = blockIdx.x * blockDim.x + threadIdx.x;  // 0..2047
  const int b   = idx >> 8;
  const int u   = idx & 255;

  float c = c0[idx];
  float n = n0v[idx];

  const size_t base = (size_t)b * T_ * U_ + u;
  const _Float16* pq = proj + base;
  const _Float16* pk = pq + MU_;
  const _Float16* pv = pq + 2 * MU_;
  const _Float16* pi = pq + 3 * MU_;
  const _Float16* pf = pq + 4 * MU_;
  const _Float16* po = pq + 5 * MU_;
  float* ho = out + base;

  __builtin_prefetch(pq, 0, 1);
  __builtin_prefetch(pk, 0, 1);
  __builtin_prefetch(pv, 0, 1);
  __builtin_prefetch(pi, 0, 1);
  __builtin_prefetch(pf, 0, 1);
  __builtin_prefetch(po, 0, 1);

  #pragma unroll 8
  for (int t = 0; t < T_; ++t) {
    const size_t off = (size_t)t * U_;
    const float q = (float)pq[off];
    const float k = (float)pk[off];
    const float v = (float)pv[off];
    const float i = (float)pi[off];
    const float f = (float)pf[off];
    const float o = (float)po[off];

    c = f * c + i * v * k;
    n = f * n + i * k;
    const float num = c * q;
    const float den = fmaxf(fabsf(n * q), 1.0f);
    ho[off] = o * (num / den);
  }
}

// ---------------------------------------------------------------------------
extern "C" void kernel_launch(void* const* d_in, const int* in_sizes, int n_in,
                              void* d_out, int out_size, void* d_ws, size_t ws_size,
                              hipStream_t stream) {
  (void)in_sizes; (void)n_in; (void)out_size; (void)ws_size;

  const float* x    = (const float*)d_in[0];  // [B, T, D]
  const float* c0   = (const float*)d_in[1];  // [B, U]
  const float* n0   = (const float*)d_in[2];  // [B, U]
  const float* W    = (const float*)d_in[3];  // [6, D, U]
  const float* bias = (const float*)d_in[4];  // [6, U]
  float* out = (float*)d_out;                 // [B, T, U]

  // Workspace layout (f16): xh [M*D] | Wt [6*U*D] | proj [6*M*U]
  _Float16* xh   = (_Float16*)d_ws;                       // 16 MiB
  _Float16* Wt   = xh + (size_t)M_ * D_;                  // 768 KiB
  _Float16* proj = Wt + (size_t)P_ * U_ * D_;             // 96 MiB

  cvt_x_f16 <<<dim3((M_ * D_ / 4) / 256), 256, 0, stream>>>(x, xh);
  cvt_wT_f16<<<dim3((P_ * D_ * U_) / 256), 256, 0, stream>>>(W, Wt);

  dim3 g1(M_ / 128, P_);                                  // (256, 6)
  xlstm_proj_wmma<<<g1, 256, LDS_BYTES, stream>>>(xh, Wt, bias, proj);

  xlstm_scan<<<dim3((B_ * U_) / 32), 32, 0, stream>>>(proj, c0, n0, out);
}
